// HyperbolicSelfAttention_8057358647460
// MI455X (gfx1250) — compile-verified, matrix operand order checked by reading, not run on hardware
//
#include <hip/hip_runtime.h>
#include <hip/hip_bf16.h>

// ---------------------------------------------------------------------------
// Hyperbolic self-attention, MI455X (gfx1250, wave32, WMMA).
// All contractions on v_wmma_f32_16x16x32_bf16, f32 accumulate.
// Operands pre-packed to bf16 so the WMMA hot loops issue pure b128 loads.
// GEMM: 64x64 register tile per wave -> 16 WMMAs per 16 b128 loads/k-step.
// ---------------------------------------------------------------------------

typedef __attribute__((ext_vector_type(16))) __bf16 v16bf;
typedef __attribute__((ext_vector_type(8)))  __bf16 v8bf;
typedef __attribute__((ext_vector_type(4)))  __bf16 v4bf;
typedef __attribute__((ext_vector_type(2)))  __bf16 v2bf;
typedef __attribute__((ext_vector_type(8)))  float  v8f;

#define B_  4
#define T_  1024
#define C_  768
#define H_  12
#define D_  64
#define C3_ 2304   // 3*C

__device__ __forceinline__ v8f wmma_bf16(v16bf a, v16bf b, v8f c) {
    return __builtin_amdgcn_wmma_f32_16x16x32_bf16(
        false, a, false, b, (short)0, c, false, false);
}

// 16x32 bf16 fragment per CDNA5 16-bit layout (ISA 7.12.2): lane-half (+8
// halfwords for lanes 16..31) already folded into p by the caller; elements
// 0..7 = p[0..7], 8..15 = p[16..23]  ->  two 16-byte loads.
__device__ __forceinline__ v16bf fragb(const __bf16* __restrict__ p) {
    v8bf lo = *(const v8bf*)p;
    v8bf hi = *(const v8bf*)(p + 16);
    v16bf f;
#pragma unroll
    for (int i = 0; i < 8; ++i) { f[i] = lo[i]; f[i + 8] = hi[i]; }
    return f;
}

// ---------------------------------------------------------------------------
// f32 -> bf16 row-major pack (x, Wqkv, Wproj).
// ---------------------------------------------------------------------------
__global__ __launch_bounds__(256) void pack_bf16(
    const float* __restrict__ in, __bf16* __restrict__ out, int n) {
    int i = (int)(blockIdx.x * blockDim.x + threadIdx.x) * 4;
    if (i >= n) return;
    float4 v = *(const float4*)(in + i);
    v4bf o;
    o[0] = (__bf16)v.x; o[1] = (__bf16)v.y;
    o[2] = (__bf16)v.z; o[3] = (__bf16)v.w;
    *(v4bf*)(out + i) = o;
}

// ---------------------------------------------------------------------------
// Out[M,N] = A[M,K] * W[N,K]^T, bf16 operands, f32 out.
// One wave per 64x64 output tile: 4 resident A-frags x 4 B-frags ->
// 16 independent WMMAs per k-step (enough ILP for the XDL pipe at low
// occupancy), 16 b128 loads per k-step, prefetch 2 k-steps ahead.
// ---------------------------------------------------------------------------
__global__ __launch_bounds__(256) void gemm_bf16_wmma(
    const __bf16* __restrict__ A, const __bf16* __restrict__ W,
    float* __restrict__ Out, int M, int N, int K) {
    const int wave = (int)((blockIdx.x * blockDim.x + threadIdx.x) >> 5);
    const int lane = threadIdx.x & 31;
    const int half = lane >> 4;
    const int l16  = lane & 15;
    const int ntn  = N >> 6;              // 64-wide N tiles
    const int tm   = wave / ntn;
    const int tn   = wave - tm * ntn;
    if (tm * 64 >= M) return;

    const __bf16* a0 = A + (size_t)(tm * 64 + l16) * K + half * 8;
    const __bf16* w0 = W + (size_t)(tn * 64 + l16) * K + half * 8;

    v8f acc[4][4];
#pragma unroll
    for (int m = 0; m < 4; ++m)
#pragma unroll
        for (int s = 0; s < 4; ++s) acc[m][s] = (v8f){};

    for (int k0 = 0; k0 < K; k0 += 32) {
        __builtin_prefetch(a0 + k0 + 64, 0, 3);   // global_prefetch_b8
        __builtin_prefetch(w0 + k0 + 64, 0, 3);
        v16bf fa[4];
#pragma unroll
        for (int m = 0; m < 4; ++m)
            fa[m] = fragb(a0 + (size_t)(m * 16) * K + k0);
#pragma unroll
        for (int s = 0; s < 4; ++s) {
            v16bf fb = fragb(w0 + (size_t)(s * 16) * K + k0);
#pragma unroll
            for (int m = 0; m < 4; ++m)
                acc[m][s] = wmma_bf16(fa[m], fb, acc[m][s]);
        }
    }
    // C/D layout: lane L, VGPR r -> row r + 8*(L>>4), col L&15
#pragma unroll
    for (int m = 0; m < 4; ++m)
#pragma unroll
        for (int s = 0; s < 4; ++s) {
            float* orow = Out + (size_t)(tm * 64 + m * 16 + 8 * half) * N
                              + tn * 64 + s * 16 + l16;
#pragma unroll
            for (int r = 0; r < 8; ++r) orow[(size_t)r * N] = acc[m][s][r];
        }
}

// ---------------------------------------------------------------------------
// Poincare-ball projection of q and k heads: one wave per row of D=64.
// Emits bf16 qp/kp (WMMA operands) + f32 squared norms.
// ---------------------------------------------------------------------------
__global__ __launch_bounds__(256) void project_qk(
    const float* __restrict__ qkv, const float* __restrict__ kcurv,
    __bf16* __restrict__ qp, __bf16* __restrict__ kp,
    float* __restrict__ x2, float* __restrict__ y2) {
    const int wid  = (int)(blockIdx.x * (blockDim.x >> 5) + (threadIdx.x >> 5));
    const int lane = threadIdx.x & 31;
    const int rows = B_ * H_ * T_;
    const int which = wid / rows;
    if (which >= 2) return;
    int rem = wid - which * rows;
    const int b = rem / (H_ * T_); rem -= b * H_ * T_;
    const int h = rem / T_;
    const int t = rem - h * T_;

    const float* src = qkv + ((size_t)b * T_ + t) * C3_ + which * C_ + h * D_;
    float2 v = *(const float2*)(src + lane * 2);
    float ss = v.x * v.x + v.y * v.y;
#pragma unroll
    for (int m = 16; m > 0; m >>= 1) ss += __shfl_xor(ss, m, 32);

    const float k = kcurv[h];
    const float maxnorm = (1.0f - 0.004f) * rsqrtf(k);
    const float ss_c = fmaxf(ss, 1e-15f);
    const float norm = sqrtf(ss_c);
    const bool  clip = norm > maxnorm;
    const float scale = clip ? (maxnorm / norm) : 1.0f;

    __bf16* dst = (which == 0 ? qp : kp) + ((size_t)(b * H_ + h) * T_ + t) * D_;
    v2bf o; o[0] = (__bf16)(v.x * scale); o[1] = (__bf16)(v.y * scale);
    *(v2bf*)(dst + lane * 2) = o;
    if (lane == 0) {
        float n2 = clip ? (maxnorm * maxnorm) : ss_c;
        (which == 0 ? x2 : y2)[(size_t)(b * H_ + h) * T_ + t] = n2;
    }
}

// ---------------------------------------------------------------------------
// V transpose + bf16 pack: qkv[b,t,2C+h*64+d] -> vT[(bh*64+d)*T + t].
// One block per (bh, 64-row t-tile); LDS-tiled so both sides are coalesced.
// ---------------------------------------------------------------------------
__global__ __launch_bounds__(256) void transpose_v_bf16(
    const float* __restrict__ qkv, __bf16* __restrict__ vT) {
    __shared__ __bf16 tile[64][72];       // [t_local][d], padded
    const int bh = (int)blockIdx.x >> 4;  // 48 heads
    const int t0 = ((int)blockIdx.x & 15) * 64;
    const int b  = bh / H_, h = bh - b * H_;
    const int tid = threadIdx.x;
#pragma unroll
    for (int r = 0; r < 16; ++r) {
        int linear = r * 256 + tid;
        int tl = linear >> 6;             // 0..63
        int d  = linear & 63;
        tile[tl][d] = (__bf16)qkv[((size_t)b * T_ + t0 + tl) * C3_
                                  + 2 * C_ + h * D_ + d];
    }
    __syncthreads();
#pragma unroll
    for (int r = 0; r < 16; ++r) {
        int linear = r * 256 + tid;
        int d  = linear >> 6;             // 0..63
        int tl = linear & 63;
        vT[((size_t)bh * D_ + d) * T_ + t0 + tl] = tile[tl][d];
    }
}

// ---------------------------------------------------------------------------
// Fused causal hyperbolic attention (flash-style; weights bounded by
// 1/WEI_EPS so no max-shift needed). One wave per (b,h, 16-row q tile).
// Per 32-key chunk: 4 WMMAs for scores, f32 distance math, wei staged in
// bf16 through per-wave LDS to re-fragment C->A layout, 4 WMMAs for wei*V.
// ---------------------------------------------------------------------------
__global__ __launch_bounds__(256) void attn_kernel(
    const __bf16* __restrict__ qp, const __bf16* __restrict__ kp,
    const float* __restrict__ x2g, const float* __restrict__ y2g,
    const __bf16* __restrict__ vT, const float* __restrict__ kcurv,
    __bf16* __restrict__ Y) {
    __shared__ __attribute__((aligned(16))) __bf16 weis[8][16][40];

    const int w    = threadIdx.x >> 5;
    const int lane = threadIdx.x & 31;
    const int half = lane >> 4;
    const int l16  = lane & 15;
    const int wid  = (int)(blockIdx.x * 8 + w);
    const int bh   = wid >> 6;
    const int ti   = wid & 63;
    if (bh >= B_ * H_) return;
    const int b = bh / H_, h = bh - b * H_;
    const int i0 = ti * 16;

    const float kc = kcurv[h];
    const float sk = sqrtf(kc);
    const float inv_sk2 = 2.0f / sk;

    const __bf16* qrow = qp + ((size_t)bh * T_ + i0 + l16) * D_ + half * 8;
    const v16bf aq0 = fragb(qrow);
    const v16bf aq1 = fragb(qrow + 32);

    float x2r[8];
#pragma unroll
    for (int r = 0; r < 8; ++r)
        x2r[r] = x2g[(size_t)bh * T_ + i0 + r + 8 * half];

    v8f acc[4];
#pragma unroll
    for (int d = 0; d < 4; ++d) acc[d] = (v8f){};
    float rowsum[8];
#pragma unroll
    for (int r = 0; r < 8; ++r) rowsum[r] = 0.0f;

    const __bf16* vthead = vT + (size_t)bh * D_ * T_;

    for (int j0 = 0; j0 <= i0 + 15; j0 += 32) {
        // ---- scores: qp_tile . kp_tile^T (contract over D=64) ----
        v8f xy[2];
#pragma unroll
        for (int kt = 0; kt < 2; ++kt) {
            const __bf16* krow =
                kp + ((size_t)bh * T_ + j0 + kt * 16 + l16) * D_ + half * 8;
            v16bf bk0 = fragb(krow);
            v16bf bk1 = fragb(krow + 32);
            v8f t = {};
            t = wmma_bf16(aq0, bk0, t);
            t = wmma_bf16(aq1, bk1, t);
            xy[kt] = t;
        }

        // ---- hyperbolic distance -> wei, causal mask, stage to LDS ----
#pragma unroll
        for (int kt = 0; kt < 2; ++kt) {
            const int   colg = j0 + kt * 16 + l16;
            const float y2c  = y2g[(size_t)bh * T_ + colg];
#pragma unroll
            for (int r = 0; r < 8; ++r) {
                const int   rowg = i0 + r + 8 * half;
                const float xyv  = xy[kt][r];
                const float x2i  = x2r[r];
                const float Aa   = 1.0f - 2.0f * kc * xyv + kc * y2c;
                const float Bb   = 1.0f - kc * x2i;
                const float num2 = Aa * Aa * x2i + Bb * Bb * y2c
                                 - 2.0f * Aa * Bb * xyv;
                const float den  = 1.0f - 2.0f * kc * xyv + kc * kc * x2i * y2c;
                const float madd = sqrtf(fmaxf(num2, 1e-15f))
                                 / fmaxf(den, 1e-15f);
                const float z    = fminf(fmaxf(sk * madd, 0.0f), 1.0f - 1e-5f);
                const float dist = inv_sk2 * 0.5f
                                 * __logf((1.0f + z) / (1.0f - z));
                float wv = 1.0f / (1e-3f + dist * dist);
                wv = (colg <= rowg) ? wv : 0.0f;
                weis[w][r + 8 * half][kt * 16 + l16] = (__bf16)wv;
                rowsum[r] += wv;
            }
        }
        // Same-wave LDS ops are in-order (DScnt) -> safe to read back.

        // ---- re-fragment wei (C-layout -> A-layout, K = 32 keys) ----
        v16bf aw = fragb(&weis[w][l16][half * 8]);

        // ---- acc += wei . V : B-frag from vT row d, contiguous in keys ----
#pragma unroll
        for (int dt = 0; dt < 4; ++dt) {
            const __bf16* vrow =
                vthead + (size_t)(dt * 16 + l16) * T_ + j0 + half * 8;
            v16bf bv = fragb(vrow);
            acc[dt] = wmma_bf16(aw, bv, acc[dt]);
        }
    }

    // Row sums: xor-reduce each 16-lane half (matches C-layout row split).
#pragma unroll
    for (int r = 0; r < 8; ++r)
#pragma unroll
        for (int m = 1; m < 16; m <<= 1)
            rowsum[r] += __shfl_xor(rowsum[r], m, 32);

    // Normalized context, written directly in bf16 for the output GEMM.
#pragma unroll
    for (int dt = 0; dt < 4; ++dt) {
        __bf16* yrow = Y + ((size_t)b * T_ + i0 + 8 * half) * C_
                         + h * D_ + dt * 16 + l16;
#pragma unroll
        for (int r = 0; r < 8; ++r)
            yrow[(size_t)r * C_] = (__bf16)(acc[dt][r] / rowsum[r]);
    }
}

// ---------------------------------------------------------------------------
// Launch
// ---------------------------------------------------------------------------
extern "C" void kernel_launch(void* const* d_in, const int* in_sizes, int n_in,
                              void* d_out, int out_size, void* d_ws, size_t ws_size,
                              hipStream_t stream) {
    const float* x     = (const float*)d_in[0];   // [B,T,C]
    const float* Wqkv  = (const float*)d_in[1];   // [3C,C]
    const float* Wproj = (const float*)d_in[2];   // [C,C]
    const float* kcurv = (const float*)d_in[3];   // [H]
    float* out = (float*)d_out;                   // [B,T,C]

    char* ws = (char*)d_ws;
    size_t o = 0;
    auto alloc_f32 = [&](size_t n) { float*  p = (float*)(ws + o);  o += n * 4; return p; };
    auto alloc_bf  = [&](size_t n) { __bf16* p = (__bf16*)(ws + o); o += n * 2; return p; };

    float*  qkv   = alloc_f32((size_t)B_ * T_ * C3_);       // 37.7 MB
    float*  x2    = alloc_f32((size_t)B_ * H_ * T_);
    float*  y2    = alloc_f32((size_t)B_ * H_ * T_);
    __bf16* x_bf  = alloc_bf((size_t)B_ * T_ * C_);
    __bf16* Wq_bf = alloc_bf((size_t)C3_ * C_);
    __bf16* Wp_bf = alloc_bf((size_t)C_ * C_);
    __bf16* qp    = alloc_bf((size_t)B_ * H_ * T_ * D_);
    __bf16* kp    = alloc_bf((size_t)B_ * H_ * T_ * D_);
    __bf16* vTb   = alloc_bf((size_t)B_ * H_ * D_ * T_);
    __bf16* Y_bf  = alloc_bf((size_t)B_ * T_ * C_);
    // total ~ 74 MB

    const int M = B_ * T_;   // 4096

    // 0) bf16 packs of GEMM operands
    {
        int n1 = M * C_, n2 = C3_ * C_, n3 = C_ * C_;
        pack_bf16<<<(n1 / 4 + 255) / 256, 256, 0, stream>>>(x,     x_bf,  n1);
        pack_bf16<<<(n2 / 4 + 255) / 256, 256, 0, stream>>>(Wqkv,  Wq_bf, n2);
        pack_bf16<<<(n3 / 4 + 255) / 256, 256, 0, stream>>>(Wproj, Wp_bf, n3);
    }
    // 1) qkv = x @ Wqkv^T   (M=4096, N=2304, K=768)
    {
        int waves = (M / 64) * (C3_ / 64);
        gemm_bf16_wmma<<<(waves * 32 + 255) / 256, 256, 0, stream>>>(
            x_bf, Wq_bf, qkv, M, C3_, C_);
    }
    // 2) projection of q/k heads, bf16 outputs + norms
    {
        int waves = 2 * B_ * H_ * T_;
        project_qk<<<(waves * 32 + 255) / 256, 256, 0, stream>>>(
            qkv, kcurv, qp, kp, x2, y2);
    }
    // 2b) V -> bf16 [B,H,D,T] transpose
    transpose_v_bf16<<<B_ * H_ * (T_ / 64), 256, 0, stream>>>(qkv, vTb);
    // 3) fused causal hyperbolic attention -> Y (bf16)
    {
        int waves = (B_ * H_) * (T_ / 16);
        attn_kernel<<<(waves + 7) / 8, 256, 0, stream>>>(
            qp, kp, x2, y2, vTb, kcurv, Y_bf);
    }
    // 4) out = Y @ Wproj^T  (M=4096, N=768, K=768)
    {
        int waves = (M / 64) * (C_ / 64);
        gemm_bf16_wmma<<<(waves * 32 + 255) / 256, 256, 0, stream>>>(
            Y_bf, Wp_bf, out, M, C_, C_);
    }
}
